// GCNlarge_12043088298517
// MI455X (gfx1250) — matrix operand order repaired and checked by reading, not compiled, not run on hardware
//
#include <hip/hip_runtime.h>
#include <hip/hip_bf16.h>
#include <stdint.h>

// ---------------- CDNA5 WMMA types ----------------
typedef __attribute__((ext_vector_type(16))) __bf16 v16bf;
typedef __attribute__((ext_vector_type(8)))  __bf16 v8bf;
typedef __attribute__((ext_vector_type(8)))  float  v8f;

#define M_NODES 20000
#define E_EDGES 100000

// round-to-nearest-even fp32 -> bf16 in integer domain
__device__ __forceinline__ unsigned short f2bf(float f) {
  unsigned u = __float_as_uint(f);
  u += 0x7FFFu + ((u >> 16) & 1u);
  return (unsigned short)(u >> 16);
}

// ---------------- degree / dinv ----------------
__global__ void k_deg_init(float* deg, int n) {
  int i = blockIdx.x * blockDim.x + threadIdx.x;
  if (i < n) deg[i] = 1.0f;              // self-loop contributes 1
}
__global__ void k_deg_acc(const long long* __restrict__ ei, float* deg, int E) {
  int i = blockIdx.x * blockDim.x + threadIdx.x;
  if (i < E) atomicAdd(&deg[(int)ei[(size_t)E + i]], 1.0f);   // dst row
}
__global__ void k_rsqrt(float* d, int n) {
  int i = blockIdx.x * blockDim.x + threadIdx.x;
  if (i < n) d[i] = rsqrtf(d[i]);
}

// ---------------- fp32 -> bf16 conversions (padded) ----------------
__global__ void k_conv_x(const float* __restrict__ x, unsigned short* __restrict__ xb,
                         int K, int Kp) {
  int k = blockIdx.x * blockDim.x + threadIdx.x;
  int m = blockIdx.y;
  if (k < Kp) xb[(size_t)m * Kp + k] = (k < K) ? f2bf(x[(size_t)m * K + k]) : (unsigned short)0;
}
// W[K x N] fp32 -> wT[Np x Kp] bf16 (transposed, zero padded)
__global__ void k_conv_wT(const float* __restrict__ W, unsigned short* __restrict__ wT,
                          int K, int N, int Kp) {
  int k = blockIdx.x * blockDim.x + threadIdx.x;
  int n = blockIdx.y;
  if (k < Kp)
    wT[(size_t)n * Kp + k] = (k < K && n < N) ? f2bf(W[(size_t)k * N + n]) : (unsigned short)0;
}

// ---------------- WMMA bf16 GEMM + fused bias/self-loop epilogue ----------------
// lin[M x Np] = A[M x Kp] * BT[Np x Kp]^T
// agg[M x Np] = bias + dinv^2 * lin            (self-loop term of GCN aggregation)
// block = 256 threads = 8 waves; wave -> 32(M) x 32(N) tile (4 WMMAs/iter);
// block -> 256M x 32N.  Grid: x = N-tiles (fast) so blocks sharing an A slab
// are adjacent in dispatch order -> A streams from HBM once, re-read hits L2.
__global__ void k_gemm(const __bf16* __restrict__ A, const __bf16* __restrict__ BT,
                       float* __restrict__ lin, float* __restrict__ agg,
                       const float* __restrict__ bias, const float* __restrict__ dinv,
                       int M, int Kp, int Np, int Nvalid) {
  const int lane = threadIdx.x & 31;
  const int wave = threadIdx.x >> 5;
  const int m0 = blockIdx.y * 256 + wave * 32;
  const int n0 = blockIdx.x * 32;
  if (m0 >= M) return;                     // M is a multiple of 32: full tiles only
  const int half = lane >> 4;              // 0: lanes 0-15, 1: lanes 16-31
  const int lr   = lane & 15;

  // A frag (16x32 bf16): lane<16 holds K 0..15; lane>=16 holds K 16..31
  const __bf16* aRow0 = A + (size_t)(m0 + lr)      * Kp + half * 16;
  const __bf16* aRow1 = A + (size_t)(m0 + 16 + lr) * Kp + half * 16;
  // B frag (32x16 bf16, transposed weights): lane<16 K 0..15, lane>=16 K 16..31
  const __bf16* bCol0 = BT + (size_t)(n0 + lr)      * Kp + half * 16;
  const __bf16* bCol1 = BT + (size_t)(n0 + 16 + lr) * Kp + half * 16;

  v8f c00 = {}, c01 = {}, c10 = {}, c11 = {};
  for (int k0 = 0; k0 < Kp; k0 += 32) {
    __builtin_prefetch(aRow0 + k0 + 256, 0, 1);   // global_prefetch_b8
    __builtin_prefetch(aRow1 + k0 + 256, 0, 1);
    v8bf a0lo = *(const v8bf*)(aRow0 + k0);
    v8bf a0hi = *(const v8bf*)(aRow0 + k0 + 8);
    v8bf a1lo = *(const v8bf*)(aRow1 + k0);
    v8bf a1hi = *(const v8bf*)(aRow1 + k0 + 8);
    v16bf a0 = __builtin_shufflevector(a0lo, a0hi, 0,1,2,3,4,5,6,7,8,9,10,11,12,13,14,15);
    v16bf a1 = __builtin_shufflevector(a1lo, a1hi, 0,1,2,3,4,5,6,7,8,9,10,11,12,13,14,15);
    v16bf b0 = *(const v16bf*)(bCol0 + k0);
    v16bf b1 = *(const v16bf*)(bCol1 + k0);
    c00 = __builtin_amdgcn_wmma_f32_16x16x32_bf16(false, a0, false, b0, (short)0, c00, false, false);
    c01 = __builtin_amdgcn_wmma_f32_16x16x32_bf16(false, a0, false, b1, (short)0, c01, false, false);
    c10 = __builtin_amdgcn_wmma_f32_16x16x32_bf16(false, a1, false, b0, (short)0, c10, false, false);
    c11 = __builtin_amdgcn_wmma_f32_16x16x32_bf16(false, a1, false, b1, (short)0, c11, false, false);
  }

  // C/D layout: VGPR r -> M = r + 8*half, N = lane&15
  const int   col0 = n0 + lr;
  const int   col1 = n0 + 16 + lr;
  const float bs0  = (col0 < Nvalid) ? bias[col0] : 0.0f;
  const float bs1  = (col1 < Nvalid) ? bias[col1] : 0.0f;
#pragma unroll
  for (int r = 0; r < 8; ++r) {
    int rowA = m0 + half * 8 + r;
    int rowB = rowA + 16;
    float dA = dinv[rowA]; float d2A = dA * dA;
    float dB = dinv[rowB]; float d2B = dB * dB;
    size_t oA = (size_t)rowA * Np;
    size_t oB = (size_t)rowB * Np;
    lin[oA + col0] = c00[r];  agg[oA + col0] = bs0 + d2A * c00[r];
    lin[oA + col1] = c01[r];  agg[oA + col1] = bs1 + d2A * c01[r];
    lin[oB + col0] = c10[r];  agg[oB + col0] = bs0 + d2B * c10[r];
    lin[oB + col1] = c11[r];  agg[oB + col1] = bs1 + d2B * c11[r];
  }
}

// ---------------- edge scatter: agg[dst] += lin[src] * dinv[src]*dinv[dst] ----------------
__global__ void k_scatter(const long long* __restrict__ ei, const float* __restrict__ lin,
                          const float* __restrict__ dinv, float* __restrict__ agg,
                          int E, int N, int Np) {
  int e = blockIdx.x;
  int src = (int)ei[e];
  int dst = (int)ei[(size_t)E + e];
  float norm = dinv[src] * dinv[dst];
  const float* ls = lin + (size_t)src * Np;
  float*       ad = agg + (size_t)dst * Np;
  for (int n = threadIdx.x; n < N; n += blockDim.x)
    atomicAdd(&ad[n], ls[n] * norm);
}

// relu + bf16 convert for next layer's A operand (pad cols stay 0)
__global__ void k_act(const float* __restrict__ agg, unsigned short* __restrict__ hb, int Np) {
  int n = blockIdx.x * blockDim.x + threadIdx.x;
  int m = blockIdx.y;
  if (n < Np) hb[(size_t)m * Np + n] = f2bf(fmaxf(agg[(size_t)m * Np + n], 0.0f));
}

// final: copy valid 2 classes (no relu on layer 4)
__global__ void k_final(const float* __restrict__ agg, float* __restrict__ out, int Np) {
  int i = blockIdx.x * blockDim.x + threadIdx.x;
  if (i < M_NODES * 2) out[i] = agg[(size_t)(i >> 1) * Np + (i & 1)];
}

// ---------------- host orchestration ----------------
extern "C" void kernel_launch(void* const* d_in, const int* in_sizes, int n_in,
                              void* d_out, int out_size, void* d_ws, size_t ws_size,
                              hipStream_t stream) {
  (void)in_sizes; (void)n_in; (void)out_size; (void)ws_size;
  const float*     x  = (const float*)d_in[0];
  const long long* ei = (const long long*)d_in[1];     // int64 edge_index [2 x E]
  const float* W[4] = {(const float*)d_in[2], (const float*)d_in[4],
                       (const float*)d_in[6], (const float*)d_in[8]};
  const float* b[4] = {(const float*)d_in[3], (const float*)d_in[5],
                       (const float*)d_in[7], (const float*)d_in[9]};

  // dims: K valid / N valid / K padded (mult 32) / N padded (mult 32)
  const int Kd[4] = {3000, 2000, 500, 100};
  const int Nd[4] = {2000,  500, 100,   2};
  const int Kp[4] = {3008, 2016, 512, 128};
  const int Np[4] = {2016,  512, 128,  32};

  char* ws = (char*)d_ws;
  size_t off = 0;
  auto carve = [&](size_t bytes) -> char* {
    char* p = ws + off;
    off += (bytes + 255) & ~(size_t)255;
    return p;
  };
  float*          dinv = (float*)carve((size_t)M_NODES * 4);            // deg -> dinv in place
  unsigned short* xb   = (unsigned short*)carve((size_t)M_NODES * 3008 * 2);
  unsigned short* wT   = (unsigned short*)carve((size_t)2016 * 3008 * 2);
  float*          lin  = (float*)carve((size_t)M_NODES * 2016 * 4);
  float*          agg  = (float*)carve((size_t)M_NODES * 2016 * 4);
  unsigned short* hb   = xb;  // alias: x-bf16 region dead after layer-1 GEMM (80.6MB <= 120.3MB)

  dim3 B(256);
  // degrees (incl self-loops) -> dinv
  k_deg_init<<<(M_NODES + 255) / 256, B, 0, stream>>>(dinv, M_NODES);
  k_deg_acc <<<(E_EDGES + 255) / 256, B, 0, stream>>>(ei, dinv, E_EDGES);
  k_rsqrt   <<<(M_NODES + 255) / 256, B, 0, stream>>>(dinv, M_NODES);

  // x -> bf16, K padded
  k_conv_x<<<dim3((3008 + 255) / 256, M_NODES), B, 0, stream>>>(x, xb, 3000, 3008);

  const unsigned short* inb = xb;
  for (int l = 0; l < 4; ++l) {
    k_conv_wT<<<dim3((Kp[l] + 255) / 256, Np[l]), B, 0, stream>>>(W[l], wT, Kd[l], Nd[l], Kp[l]);
    // grid.x = N tiles (fast-varying), grid.y = M tiles -> A-slab reuse in L2
    k_gemm<<<dim3(Np[l] / 32, (M_NODES + 255) / 256), B, 0, stream>>>(
        (const __bf16*)inb, (const __bf16*)wT, lin, agg, b[l], dinv,
        M_NODES, Kp[l], Np[l], Nd[l]);
    int tb = (Nd[l] <= 64) ? 64 : (Nd[l] <= 128 ? 128 : 256);
    k_scatter<<<E_EDGES, dim3(tb), 0, stream>>>(ei, lin, dinv, agg, E_EDGES, Nd[l], Np[l]);
    if (l < 3) {
      k_act<<<dim3((Np[l] + 255) / 256, M_NODES), B, 0, stream>>>(agg, hb, Np[l]);
      inb = hb;   // lead dim Np[l] == Kp[l+1]
    }
  }
  k_final<<<(M_NODES * 2 + 255) / 256, B, 0, stream>>>(agg, (float*)d_out, 32);
}
